// WindowAttention_23235773071538
// MI455X (gfx1250) — compile-verified, matrix operand order checked by reading, not run on hardware
//
#include <hip/hip_runtime.h>
#include <hip/hip_bf16.h>

// ---------------------------------------------------------------------------
// Window attention (Swin-style), MI455X / gfx1250, wave32 + WMMA bf16 +
// async global->LDS staging (ASYNCcnt).
// H=W=56 divisible by WS=7 -> no padding; window partition is a permutation
// folded into GEMM addressing.
// ---------------------------------------------------------------------------

typedef __attribute__((ext_vector_type(16))) __bf16 v16bf;
typedef __attribute__((ext_vector_type(8)))  float  v8f;
typedef __attribute__((ext_vector_type(4)))  int    v4i;

#define C_DIM   512
#define NHEAD   16
#define HD      32
#define WS7     7
#define NTOK    49            // tokens per window
#define NWIN    1024          // 16 * 8 * 8
#define MTOT    (NWIN * NTOK) // 50176 tokens total (multiple of 16)
#define MAT_ELEMS ((size_t)NWIN * NHEAD * NTOK * HD) // 25,690,112 bf16 per matrix
#define SCALE_F 0.17677669529663687f                  // 32^-0.5

#if defined(__has_builtin)
#if __has_builtin(__builtin_amdgcn_global_load_async_to_lds_b128)
#define HAVE_ASYNC_LDS 1
#endif
#endif
#ifndef HAVE_ASYNC_LDS
#define HAVE_ASYNC_LDS 0
#endif

#define AS1 __attribute__((address_space(1)))
#define AS3 __attribute__((address_space(3)))

// copy 16 bytes global -> LDS (async on gfx1250, sync fallback otherwise)
static __device__ __forceinline__ void async_cp16(const void* g, void* l) {
#if HAVE_ASYNC_LDS
    __builtin_amdgcn_global_load_async_to_lds_b128(
        (AS1 v4i*)(void*)g, (AS3 v4i*)l, 0, 0);
#else
    *(v4i*)l = *(const v4i*)g;
#endif
}

static __device__ __forceinline__ void wait_async() {
#if HAVE_ASYNC_LDS
#if __has_builtin(__builtin_amdgcn_s_wait_asynccnt)
    __builtin_amdgcn_s_wait_asynccnt(0);
#else
    asm volatile("s_wait_asynccnt 0x0" ::: "memory");
#endif
#endif
}

static __device__ __forceinline__ __bf16 f2bf(float f) {
    unsigned u = __float_as_uint(f);
    unsigned r = (u + 0x7fffu + ((u >> 16) & 1u)) >> 16;   // RNE
    unsigned short s = (unsigned short)r;
    return __builtin_bit_cast(__bf16, s);
}

// token index m in window order -> source row in [B, H*W] flat token space
static __device__ __forceinline__ int src_row(int m) {
    int win = m / NTOK, tok = m - win * NTOK;
    int b  = win >> 6;          // 64 windows per image
    int wq = win & 63;
    int wr = wq >> 3, wc = wq & 7;
    int tr = tok / WS7, tc = tok - tr * WS7;
    return b * 3136 + (wr * WS7 + tr) * 56 + (wc * WS7 + tc);
}

// ---------------------------------------------------------------------------
// Kernel 1: fp32 -> bf16 weight conversion (Wq, Wk, Wv, Wp -> ws)
// ---------------------------------------------------------------------------
__global__ __launch_bounds__(256)
void convert_weights(const float* __restrict__ Wq, const float* __restrict__ Wk,
                     const float* __restrict__ Wv, const float* __restrict__ Wp,
                     __bf16* __restrict__ out) {
    int idx = blockIdx.x * 256 + threadIdx.x;      // 0 .. 4*512*512-1
    int which = idx >> 18;                          // 262144 per matrix
    int off   = idx & 0x3ffff;
    const float* src = (which == 0) ? Wq : (which == 1) ? Wk : (which == 2) ? Wv : Wp;
    out[idx] = f2bf(src[off]);
}

// ---------------------------------------------------------------------------
// Kernel 2: fused QKV projection. Y = X @ W^T + b  (then *SCALE for q),
// written bf16 into [mat][win][head][tok][hd].
// Block = 8 waves sharing one 64-column W slice (64 KB) staged async in LDS;
// wave w computes a 16(M) x 64(N) tile (M-tiles mblock*8+w).
// ---------------------------------------------------------------------------
__global__ __launch_bounds__(256)
void qkv_proj(const float* __restrict__ q, const float* __restrict__ k,
              const float* __restrict__ v,
              const __bf16* __restrict__ Wbf,    // [3][512][512] (Wq,Wk,Wv)
              const float* __restrict__ bq, const float* __restrict__ bk,
              const float* __restrict__ bv,
              __bf16* __restrict__ out) {        // [3][NWIN][NHEAD][NTOK][HD]
    __shared__ __align__(16) __bf16 Wsh[64 * C_DIM];   // 64 KB

    int tid  = threadIdx.x;
    int lane = tid & 31;
    int wv   = tid >> 5;

    int blk    = blockIdx.x;        // 0 .. 9407
    int mat    = blk / 3136;        // 3136 blocks per matrix
    int r2     = blk - mat * 3136;
    int ngrp   = r2 / 392;          // 8 N-groups of 64
    int mblock = r2 - ngrp * 392;   // 392 M-blocks of 128 rows

    const float* X    = (mat == 0) ? q  : (mat == 1) ? k  : v;
    const float* bias = (mat == 0) ? bq : (mat == 1) ? bk : bv;
    const __bf16* W   = Wbf + (size_t)mat * C_DIM * C_DIM;

    int baseN = ngrp * 64;
    int baseM = (mblock * 8 + wv) * 16;

    // stage 64 contiguous W rows (cols of W^T) into LDS: 64*512 bf16 = 64 KB
    {
        const __bf16* wsrc = W + (size_t)baseN * C_DIM;
        for (int c = tid; c < 4096; c += 256)           // 16 B chunks
            async_cp16(wsrc + c * 8, Wsh + c * 8);
        wait_async();
    }
    __syncthreads();

    // A fragment addressing: lane m (0..15) holds row m, K chunks {0..7,16..23};
    // lane m+16 holds halves {8..15,24..31}.
    int rowA  = baseM + (lane & 15);
    const float* xrow = X + (size_t)src_row(rowA) * C_DIM;
    int khalf = (lane >> 4) * 8;
    int colb  = lane & 15;

    v8f acc[4] = {};
    for (int k0 = 0; k0 < C_DIM; k0 += 32) {
        int kk = k0 + khalf;
        __builtin_prefetch(xrow + kk + 64, 0, 0);
        v16bf a;
        #pragma unroll
        for (int j = 0; j < 8; ++j) {
            a[j]     = f2bf(xrow[kk + j]);
            a[8 + j] = f2bf(xrow[kk + 16 + j]);
        }
        #pragma unroll
        for (int t = 0; t < 4; ++t) {
            const __bf16* wcol = Wsh + (size_t)(t * 16 + colb) * C_DIM;
            v16bf b;
            #pragma unroll
            for (int j = 0; j < 8; ++j) {
                b[j]     = wcol[kk + j];
                b[8 + j] = wcol[kk + 16 + j];
            }
            acc[t] = __builtin_amdgcn_wmma_f32_16x16x32_bf16(
                false, a, false, b, (short)0, acc[t], false, false);
        }
    }

    // Epilogue: C/D layout -> VGPR r holds M=r (lanes 0-15) / M=r+8 (lanes 16-31)
    float scale = (mat == 0) ? SCALE_F : 1.0f;
    int mrow = baseM + 8 * (lane >> 4);
    int winr[8], tokr[8];
    #pragma unroll
    for (int r = 0; r < 8; ++r) {
        int m = mrow + r;
        winr[r] = m / NTOK;
        tokr[r] = m - winr[r] * NTOK;
    }
    #pragma unroll
    for (int t = 0; t < 4; ++t) {
        int n = baseN + t * 16 + colb;
        float bb = bias[n];
        int head = n >> 5, hd = n & 31;
        #pragma unroll
        for (int r = 0; r < 8; ++r) {
            float val = (acc[t][r] + bb) * scale;
            size_t idx = ((((size_t)mat * NWIN + winr[r]) * NHEAD + head) * NTOK + tokr[r]) * HD + hd;
            out[idx] = f2bf(val);
        }
    }
}

// ---------------------------------------------------------------------------
// Kernel 3: per-(window, head) attention. 4 waves / block.
//   q/k/v (49x32 bf16) async-staged to LDS, zero-padded to 64 rows
//   S = q @ k^T   (4x4 WMMA tiles, K = HD = 32)
//   softmax fp32 over valid 49 columns, P bf16 in LDS
//   out = P @ V   (2 K-steps x 2 N-tiles of WMMA)
// ---------------------------------------------------------------------------
__global__ __launch_bounds__(128)
void window_attn(const __bf16* __restrict__ QKV,   // [3][NWIN][NHEAD][NTOK][HD]
                 __bf16* __restrict__ AO) {        // [NWIN][NTOK][C]
    __shared__ __align__(16) float  Ssh[64 * 64];
    __shared__ __align__(16) __bf16 Psh[64 * 64];
    __shared__ __align__(16) __bf16 Qsh[64 * HD];
    __shared__ __align__(16) __bf16 Ksh[64 * HD];
    __shared__ __align__(16) __bf16 Vsh[64 * HD];

    int wh   = blockIdx.x;          // 0 .. 16383
    int win  = wh >> 4;
    int head = wh & 15;
    int tid  = threadIdx.x;
    int wv   = tid >> 5;            // wave 0..3
    int lane = tid & 31;

    const __bf16* qb = QKV + ((size_t)win * NHEAD + head) * (NTOK * HD);
    const __bf16* kb = qb + MAT_ELEMS;
    const __bf16* vb = qb + 2 * MAT_ELEMS;

    const __bf16 ZB = __builtin_bit_cast(__bf16, (unsigned short)0);

    // async-stage q/k/v: 49*32 bf16 = 3136 B = 196 x 16 B each
    for (int c = tid; c < 196; c += 128) {
        async_cp16(qb + c * 8, Qsh + c * 8);
        async_cp16(kb + c * 8, Ksh + c * 8);
        async_cp16(vb + c * 8, Vsh + c * 8);
    }
    // zero-pad rows 49..63 (elements 1568..2047)
    for (int i = NTOK * HD + tid; i < 64 * HD; i += 128) {
        Qsh[i] = ZB; Ksh[i] = ZB; Vsh[i] = ZB;
    }
    wait_async();
    __syncthreads();

    // ---- S = q @ k^T : wave wv owns row-tile wv ----
    int khalf = (lane >> 4) * 8;
    int rowA  = wv * 16 + (lane & 15);
    v16bf a;
    #pragma unroll
    for (int j = 0; j < 8; ++j) {
        a[j]     = Qsh[rowA * HD + khalf + j];
        a[8 + j] = Qsh[rowA * HD + khalf + 16 + j];
    }
    v8f s[4];
    #pragma unroll
    for (int t = 0; t < 4; ++t) {
        int col = t * 16 + (lane & 15);
        v16bf b;                       // B[d][n] = k[n][d]
        #pragma unroll
        for (int j = 0; j < 8; ++j) {
            b[j]     = Ksh[col * HD + khalf + j];
            b[8 + j] = Ksh[col * HD + khalf + 16 + j];
        }
        v8f cz = {};
        s[t] = __builtin_amdgcn_wmma_f32_16x16x32_bf16(
            false, a, false, b, (short)0, cz, false, false);
    }
    // scatter S tiles to LDS
    int mrow = wv * 16 + 8 * (lane >> 4);
    #pragma unroll
    for (int t = 0; t < 4; ++t)
        #pragma unroll
        for (int r = 0; r < 8; ++r)
            Ssh[(mrow + r) * 64 + t * 16 + (lane & 15)] = s[t][r];
    __syncthreads();

    // ---- softmax (rows 0..48 valid; pad columns masked, pad rows zeroed) ----
    if (tid < 64) {
        int r = tid;
        if (r < NTOK) {
            float mx = -1e30f;
            for (int n = 0; n < NTOK; ++n) mx = fmaxf(mx, Ssh[r * 64 + n]);
            float sum = 0.f;
            for (int n = 0; n < NTOK; ++n) {
                float e = __expf(Ssh[r * 64 + n] - mx);
                Ssh[r * 64 + n] = e;
                sum += e;
            }
            float inv = 1.0f / sum;
            for (int n = 0; n < 64; ++n)
                Psh[r * 64 + n] = (n < NTOK) ? f2bf(Ssh[r * 64 + n] * inv) : ZB;
        } else {
            for (int n = 0; n < 64; ++n) Psh[r * 64 + n] = ZB;
        }
    }
    __syncthreads();

    // ---- out = P @ V : wave wv owns row-tile wv, N = 32 ----
    v8f o[2] = {};
    #pragma unroll
    for (int ks = 0; ks < 64; ks += 32) {
        int kk = ks + khalf;
        int row = wv * 16 + (lane & 15);
        v16bf pa;
        #pragma unroll
        for (int j = 0; j < 8; ++j) {
            pa[j]     = Psh[row * 64 + kk + j];
            pa[8 + j] = Psh[row * 64 + kk + 16 + j];
        }
        #pragma unroll
        for (int t = 0; t < 2; ++t) {
            int col = t * 16 + (lane & 15);
            v16bf vbf;                 // B[ktok][n] = V[ktok][n]
            #pragma unroll
            for (int j = 0; j < 8; ++j) {
                vbf[j]     = Vsh[(kk + j) * HD + col];
                vbf[8 + j] = Vsh[(kk + 16 + j) * HD + col];
            }
            o[t] = __builtin_amdgcn_wmma_f32_16x16x32_bf16(
                false, pa, false, vbf, (short)0, o[t], false, false);
        }
    }
    // store rows < 49 into [win][tok][C] with column = head*32 + n
    #pragma unroll
    for (int t = 0; t < 2; ++t) {
        int nloc = t * 16 + (lane & 15);
        #pragma unroll
        for (int r = 0; r < 8; ++r) {
            int m = mrow + r;
            if (m < NTOK) {
                size_t idx = ((size_t)win * NTOK + m) * C_DIM + head * HD + nloc;
                AO[idx] = f2bf(o[t][r]);
            }
        }
    }
}

// ---------------------------------------------------------------------------
// Kernel 4: output projection Y = A @ Wp^T + bp, fp32 result scattered back
// through the reverse window permutation into d_out. Same LDS W-slice scheme.
// ---------------------------------------------------------------------------
__global__ __launch_bounds__(256)
void out_proj(const __bf16* __restrict__ A,      // [MTOT][512] window order
              const __bf16* __restrict__ Wpb,    // [512][512] bf16
              const float* __restrict__ bp,
              float* __restrict__ out) {         // [B*3136][512]
    __shared__ __align__(16) __bf16 Wsh[64 * C_DIM];   // 64 KB

    int tid  = threadIdx.x;
    int lane = tid & 31;
    int wv   = tid >> 5;

    int blk    = blockIdx.x;        // 0 .. 3135
    int ngrp   = blk / 392;
    int mblock = blk - ngrp * 392;

    int baseN = ngrp * 64;
    int baseM = (mblock * 8 + wv) * 16;

    {
        const __bf16* wsrc = Wpb + (size_t)baseN * C_DIM;
        for (int c = tid; c < 4096; c += 256)
            async_cp16(wsrc + c * 8, Wsh + c * 8);
        wait_async();
    }
    __syncthreads();

    int rowA = baseM + (lane & 15);
    const __bf16* arow = A + (size_t)rowA * C_DIM;
    int khalf = (lane >> 4) * 8;
    int colb  = lane & 15;

    v8f acc[4] = {};
    for (int k0 = 0; k0 < C_DIM; k0 += 32) {
        int kk = k0 + khalf;
        __builtin_prefetch(arow + kk + 128, 0, 0);
        v16bf a;
        #pragma unroll
        for (int j = 0; j < 8; ++j) {
            a[j]     = arow[kk + j];
            a[8 + j] = arow[kk + 16 + j];
        }
        #pragma unroll
        for (int t = 0; t < 4; ++t) {
            const __bf16* wcol = Wsh + (size_t)(t * 16 + colb) * C_DIM;
            v16bf b;
            #pragma unroll
            for (int j = 0; j < 8; ++j) {
                b[j]     = wcol[kk + j];
                b[8 + j] = wcol[kk + 16 + j];
            }
            acc[t] = __builtin_amdgcn_wmma_f32_16x16x32_bf16(
                false, a, false, b, (short)0, acc[t], false, false);
        }
    }

    int mrow = baseM + 8 * (lane >> 4);
    int dstr[8];
    #pragma unroll
    for (int r = 0; r < 8; ++r) dstr[r] = src_row(mrow + r);
    #pragma unroll
    for (int t = 0; t < 4; ++t) {
        int n = baseN + t * 16 + colb;
        float bb = bp[n];
        #pragma unroll
        for (int r = 0; r < 8; ++r)
            out[(size_t)dstr[r] * C_DIM + n] = acc[t][r] + bb;
    }
}

// ---------------------------------------------------------------------------
// Launch
// ---------------------------------------------------------------------------
extern "C" void kernel_launch(void* const* d_in, const int* in_sizes, int n_in,
                              void* d_out, int out_size, void* d_ws, size_t ws_size,
                              hipStream_t stream) {
    const float* q  = (const float*)d_in[0];
    const float* k  = (const float*)d_in[1];
    const float* v  = (const float*)d_in[2];
    const float* Wq = (const float*)d_in[3];
    const float* bq = (const float*)d_in[4];
    const float* Wk = (const float*)d_in[5];
    const float* bk = (const float*)d_in[6];
    const float* Wv = (const float*)d_in[7];
    const float* bv = (const float*)d_in[8];
    const float* Wp = (const float*)d_in[9];
    const float* bp = (const float*)d_in[10];
    float* out = (float*)d_out;

    // workspace carve-up (bf16 elements)
    __bf16* Wbf = (__bf16*)d_ws;                     // 4 * 512 * 512
    __bf16* QKV = Wbf + (size_t)4 * C_DIM * C_DIM;   // 3 * MAT_ELEMS
    __bf16* AO  = QKV + 3 * MAT_ELEMS;               // MAT_ELEMS  ([win][tok][C])

    convert_weights<<<(4 * C_DIM * C_DIM) / 256, 256, 0, stream>>>(Wq, Wk, Wv, Wp, Wbf);

    // 3 mats x 8 ngroups x 392 mblocks
    qkv_proj<<<3 * 8 * 392, 256, 0, stream>>>(q, k, v, Wbf, bq, bk, bv, QKV);

    window_attn<<<NWIN * NHEAD, 128, 0, stream>>>(QKV, AO);

    // 8 ngroups x 392 mblocks
    out_proj<<<8 * 392, 256, 0, stream>>>(AO, Wbf + (size_t)3 * C_DIM * C_DIM, bp, out);
}
